// BanzhafModule_76132590289145
// MI455X (gfx1250) — matrix operand order repaired and checked by reading, not compiled, not run on hardware
//
#include <hip/hip_runtime.h>
#include <hip/hip_bf16.h>

typedef __attribute__((ext_vector_type(16))) _Float16 v16h;
typedef __attribute__((ext_vector_type(8)))  _Float16 v8h;
typedef __attribute__((ext_vector_type(8)))  float    v8f;
typedef __attribute__((ext_vector_type(4)))  int      v4i;

#ifndef __has_builtin
#define __has_builtin(x) 0
#endif
#if __has_builtin(__builtin_amdgcn_global_load_async_to_lds_b128) && \
    __has_builtin(__builtin_amdgcn_s_wait_asynccnt)
#define USE_ASYNC_LDS 1
#else
#define USE_ASYNC_LDS 0
#endif

#if USE_ASYNC_LDS
typedef __attribute__((address_space(1))) v4i* gv4i_p;   // global int4*
typedef __attribute__((address_space(3))) v4i* sv4i_p;   // LDS int4*
#endif

// ---------------------------------------------------------------------------
// Fragment loaders (per CDNA5 ISA 7.12.2 VGPR layouts, wave32)
// A 16x32 f16: lanes 0-15: M=lane, K in {0..7,16..23}; lanes 16-31: M=lane-16,
//              K in {8..15,24..31}  -> two 8-half (16B) LDS reads at +0, +16.
// B 32x16 f16: lanes 0-15: N=lane, K in {0..7,16..23}; lanes 16-31: N=lane-16,
//              K in {8..15,24..31}  -> two 8-half (16B) LDS reads at +0, +16.
// ---------------------------------------------------------------------------
__device__ inline v16h load_fragA(const _Float16* base, int row0, int lane) {
    int r  = row0 + (lane & 15);
    int kb = (lane >> 4) << 3;           // 0 or 8
    const _Float16* p = base + r * 32 + kb;
    v8h lo = *(const v8h*)p;
    v8h hi = *(const v8h*)(p + 16);
    v16h out;
#pragma unroll
    for (int i = 0; i < 8; ++i) { out[i] = lo[i]; out[i + 8] = hi[i]; }
    return out;
}

__device__ inline v16h load_fragB(const _Float16* base, int row0, int lane) {
    int r  = row0 + (lane & 15);
    int kb = (lane >> 4) << 3;           // 0 or 8
    const _Float16* p = base + r * 32 + kb;
    v8h lo = *(const v8h*)p;
    v8h hi = *(const v8h*)(p + 16);
    v16h out;
#pragma unroll
    for (int i = 0; i < 8; ++i) { out[i] = lo[i]; out[i + 8] = hi[i]; }
    return out;
}

// ---------------------------------------------------------------------------
// Batched GEMM:  C[b] = A[b] (MxK, row pitch lda) x Bt[b]^T (Bt is NxK, pitch
// ldb), C row pitch ldc.  Block tile 128x128 (grid.x = N/128, grid.y = M/128),
// 8 waves (4Mx2N), each wave owns a 32x64 tile = 2x4 f16 WMMA fragments.
// Double-buffered LDS; next tile staged while current tile computes
// (CDNA5 async-to-LDS when the builtin exists, else reg-staged loads).
// ---------------------------------------------------------------------------
template <typename OutT>
__global__ __launch_bounds__(256) void gemm_wmma(
    const _Float16* __restrict__ A, const _Float16* __restrict__ Bt,
    OutT* __restrict__ C, int K, int lda, int ldb, int ldc,
    long long strideA, long long strideB, long long strideC)
{
    const int batch = blockIdx.z;
    A  += (size_t)batch * strideA;
    Bt += (size_t)batch * strideB;
    C  += (size_t)batch * strideC;

    const int m0   = blockIdx.y * 128;
    const int n0   = blockIdx.x * 128;
    const int tid  = threadIdx.x;
    const int lane = tid & 31;
    const int wave = tid >> 5;
    const int waveM = wave & 3;   // 0..3 -> 32-row slice
    const int waveN = wave >> 2;  // 0..1 -> 64-col slice

    __shared__ __align__(16) _Float16 sA[2][128 * 32];
    __shared__ __align__(16) _Float16 sB[2][128 * 32];

    // staging map: 256 threads x 16 halfs cover a 128x32 tile
    const int sr = tid >> 1;              // 0..127 tile row
    const int sc = (tid & 1) << 4;        // 0 or 16 halfs
    const _Float16* aRow = A  + (size_t)(m0 + sr) * lda + sc;
    const _Float16* bRow = Bt + (size_t)(n0 + sr) * ldb + sc;
    _Float16* sAd = &sA[0][0] + sr * 32 + sc;
    _Float16* sBd = &sB[0][0] + sr * 32 + sc;
    const int bufHalfs = 128 * 32;

    const int nk = K >> 5;                // K / 32 steps
    v8f acc[2][4] = {};
#if !USE_ASYNC_LDS
    uint4 ra0, ra1, rb0, rb1;             // reg staging (sync fallback)
#endif

#if USE_ASYNC_LDS
#define STAGE_ISSUE(t, buf)                                                      \
    do {                                                                         \
        const _Float16* as_ = aRow + ((t) << 5);                                 \
        const _Float16* bs_ = bRow + ((t) << 5);                                 \
        _Float16* ad_ = sAd + (buf) * bufHalfs;                                  \
        _Float16* bd_ = sBd + (buf) * bufHalfs;                                  \
        __builtin_amdgcn_global_load_async_to_lds_b128(                          \
            (gv4i_p)(void*)as_,       (sv4i_p)ad_,       0, 0);                  \
        __builtin_amdgcn_global_load_async_to_lds_b128(                          \
            (gv4i_p)(void*)(as_ + 8), (sv4i_p)(ad_ + 8), 0, 0);                  \
        __builtin_amdgcn_global_load_async_to_lds_b128(                          \
            (gv4i_p)(void*)bs_,       (sv4i_p)bd_,       0, 0);                  \
        __builtin_amdgcn_global_load_async_to_lds_b128(                          \
            (gv4i_p)(void*)(bs_ + 8), (sv4i_p)(bd_ + 8), 0, 0);                  \
    } while (0)
#define STAGE_LAND(buf) __builtin_amdgcn_s_wait_asynccnt(0)
#else
#define STAGE_ISSUE(t, buf)                                                      \
    do {                                                                         \
        const _Float16* as_ = aRow + ((t) << 5);                                 \
        const _Float16* bs_ = bRow + ((t) << 5);                                 \
        ra0 = *(const uint4*)as_;  ra1 = *(const uint4*)(as_ + 8);               \
        rb0 = *(const uint4*)bs_;  rb1 = *(const uint4*)(bs_ + 8);               \
        __builtin_prefetch(as_ + 64, 0, 3);                                      \
        __builtin_prefetch(bs_ + 64, 0, 3);                                      \
    } while (0)
#define STAGE_LAND(buf)                                                          \
    do {                                                                         \
        *(uint4*)(sAd + (buf) * bufHalfs)     = ra0;                             \
        *(uint4*)(sAd + (buf) * bufHalfs + 8) = ra1;                             \
        *(uint4*)(sBd + (buf) * bufHalfs)     = rb0;                             \
        *(uint4*)(sBd + (buf) * bufHalfs + 8) = rb1;                             \
    } while (0)
#endif

    // prologue: tile 0 into buffer 0
    STAGE_ISSUE(0, 0);
    STAGE_LAND(0);
    __syncthreads();

    for (int t = 0; t < nk; ++t) {
        const int cur = t & 1;
        const int nxt = cur ^ 1;
        const bool more = (t + 1) < nk;
        if (more) STAGE_ISSUE(t + 1, nxt);

        const _Float16* cA = &sA[cur][0];
        const _Float16* cB = &sB[cur][0];
        v16h af[2], bf[4];
        af[0] = load_fragA(cA, waveM * 32,      lane);
        af[1] = load_fragA(cA, waveM * 32 + 16, lane);
#pragma unroll
        for (int j = 0; j < 4; ++j)
            bf[j] = load_fragB(cB, waveN * 64 + j * 16, lane);

#pragma unroll
        for (int i = 0; i < 2; ++i)
#pragma unroll
            for (int j = 0; j < 4; ++j)
                acc[i][j] = __builtin_amdgcn_wmma_f32_16x16x32_f16(
                    false, af[i], false, bf[j], (short)0, acc[i][j], false, false);

        if (more) STAGE_LAND(nxt);
        __syncthreads();
    }

    // C/D layout: VGPR r: lanes 0-15 -> (M=r, N=lane); lanes 16-31 -> (M=r+8)
    const int ln = lane & 15;
    const int lm = (lane >> 4) << 3;
#pragma unroll
    for (int i = 0; i < 2; ++i)
#pragma unroll
        for (int j = 0; j < 4; ++j) {
            int mb = m0 + waveM * 32 + i * 16 + lm;
            int nn = n0 + waveN * 64 + j * 16 + ln;
#pragma unroll
            for (int r = 0; r < 8; ++r)
                C[(size_t)(mb + r) * ldc + nn] = (OutT)acc[i][j][r];
        }
#undef STAGE_ISSUE
#undef STAGE_LAND
}

// ---------------------------------------------------------------------------
// Q/K/V f32 [d][n] -> f16 transposed [n][d] (WMMA B-operand layout)
// ---------------------------------------------------------------------------
__global__ void transpose_to_f16(const float* __restrict__ Q,
                                 const float* __restrict__ Kw,
                                 const float* __restrict__ V,
                                 _Float16* __restrict__ Wt)
{
    int idx = blockIdx.x * 256 + threadIdx.x;     // n*512 + d
    int w   = blockIdx.y;                         // 0=Q 1=K 2=V
    const float* src = (w == 0) ? Q : (w == 1) ? Kw : V;
    int n = idx >> 9, d = idx & 511;
    Wt[((size_t)w << 18) + idx] = (_Float16)src[((size_t)d << 9) + n];
}

// ---------------------------------------------------------------------------
// conv1 (1->512, 3x3, SAME) + ReLU.  One block per output pixel.
// Writes h (f32, [B][512][32][32]) and hx (f16, [B][1024][512]).
// ---------------------------------------------------------------------------
__global__ __launch_bounds__(256) void conv1_relu(
    const float* __restrict__ x, const float* __restrict__ W1,
    const float* __restrict__ b1, float* __restrict__ h,
    _Float16* __restrict__ hx)
{
    const int pix = blockIdx.x;            // b*1024 + t*32 + v
    const int b = pix >> 10;
    const int l = pix & 1023;
    const int t = l >> 5, v = l & 31;
    const int tid = threadIdx.x;

    __shared__ float patch[9];
    if (tid < 9) {
        int dy = tid / 3 - 1, dx = tid % 3 - 1;
        int tt = t + dy, vv = v + dx;
        float val = 0.f;
        if (tt >= 0 && tt < 32 && vv >= 0 && vv < 32)
            val = x[((size_t)b << 10) + (tt << 5) + vv];
        patch[tid] = val;
    }
    __syncthreads();

    for (int p = tid; p < 512; p += 256) {
        float acc = b1[p];
#pragma unroll
        for (int i = 0; i < 9; ++i) acc += patch[i] * W1[p * 9 + i];
        acc = fmaxf(acc, 0.f);
        h[((size_t)(b * 512 + p) << 10) + l] = acc;
        hx[((size_t)(b * 1024 + l) << 9) + p] = (_Float16)acc;
    }
}

// ---------------------------------------------------------------------------
// Row softmax over S (f32, 1024 per row); writes P as f16 IN PLACE
// (row r occupies halfs [r*2048, r*2048+1024) of the same buffer).
// ---------------------------------------------------------------------------
__global__ __launch_bounds__(256) void softmax_rows(float* __restrict__ S)
{
    __shared__ float buf[1024];
    __shared__ float red[256];
    const size_t rowOff = (size_t)blockIdx.x * 1024;
    float* row = S + rowOff;
    const int tid = threadIdx.x;

    float lmax = -3.0e38f;
    for (int i = tid; i < 1024; i += 256) { float xv = row[i]; buf[i] = xv; lmax = fmaxf(lmax, xv); }
    red[tid] = lmax; __syncthreads();
    for (int s = 128; s > 0; s >>= 1) { if (tid < s) red[tid] = fmaxf(red[tid], red[tid + s]); __syncthreads(); }
    const float m = red[0]; __syncthreads();

    float lsum = 0.f;
    for (int i = tid; i < 1024; i += 256) { float e = __expf(buf[i] - m); buf[i] = e; lsum += e; }
    red[tid] = lsum; __syncthreads();
    for (int s = 128; s > 0; s >>= 1) { if (tid < s) red[tid] += red[tid + s]; __syncthreads(); }
    const float inv = 1.0f / red[0]; __syncthreads();

    _Float16* prow = (_Float16*)S + rowOff * 2;   // pitch 2048 halfs
    for (int i = tid; i < 1024; i += 256) prow[i] = (_Float16)(buf[i] * inv);
}

// ---------------------------------------------------------------------------
// conv2 (512->1, 3x3, SAME) on (h + o).  h is [B][512][32][32] f32,
// o is [B][1024][512] f32 (token-major). One block per output pixel,
// channel-parallel partial sums + LDS tree reduction.
// ---------------------------------------------------------------------------
__global__ __launch_bounds__(256) void conv2_out(
    const float* __restrict__ h, const float* __restrict__ o,
    const float* __restrict__ W2, const float* __restrict__ b2,
    float* __restrict__ out)
{
    const int pix = blockIdx.x;            // b*1024 + t*32 + v
    const int b = pix >> 10;
    const int l = pix & 1023;
    const int t = l >> 5, v = l & 31;
    const int tid = threadIdx.x;

    float partial = 0.f;
    for (int p = tid; p < 512; p += 256) {
#pragma unroll
        for (int ky = 0; ky < 3; ++ky) {
            int tt = t + ky - 1;
            if (tt < 0 || tt >= 32) continue;
#pragma unroll
            for (int kx = 0; kx < 3; ++kx) {
                int vv = v + kx - 1;
                if (vv < 0 || vv >= 32) continue;
                int ll = (tt << 5) + vv;
                float hv = h[((size_t)(b * 512 + p) << 10) + ll];
                float ov = o[((size_t)(b * 1024 + ll) << 9) + p];
                partial += (hv + ov) * W2[p * 9 + ky * 3 + kx];
            }
        }
    }
    __shared__ float red[256];
    red[tid] = partial; __syncthreads();
    for (int s = 128; s > 0; s >>= 1) { if (tid < s) red[tid] += red[tid + s]; __syncthreads(); }
    if (tid == 0) out[pix] = red[0] + b2[0];
}

// ---------------------------------------------------------------------------
// Launch: conv1 -> (q,k,vT) projections -> S=q.k^T -> softmax -> o=P.v -> conv2
// ---------------------------------------------------------------------------
extern "C" void kernel_launch(void* const* d_in, const int* in_sizes, int n_in,
                              void* d_out, int out_size, void* d_ws, size_t ws_size,
                              hipStream_t stream) {
    const float* x  = (const float*)d_in[0];
    const float* W1 = (const float*)d_in[1];
    const float* b1 = (const float*)d_in[2];
    const float* Qw = (const float*)d_in[3];
    const float* Kw = (const float*)d_in[4];
    const float* Vw = (const float*)d_in[5];
    const float* W2 = (const float*)d_in[6];
    const float* b2 = (const float*)d_in[7];
    float* outp = (float*)d_out;

    char* ws = (char*)d_ws;
    float*    h    = (float*)(ws);                       //  64 MB  f32 [32][512][32][32]
    _Float16* hx   = (_Float16*)(ws + 67108864ull);      //  32 MB  f16 [32][1024][512]
    _Float16* Wt   = (_Float16*)(ws + 100663296ull);     // 1.5 MB  f16 Qt|Kt|Vt [512][512]
    _Float16* q16  = (_Float16*)(ws + 102236160ull);     //  32 MB  f16 [32][1024][512]
    _Float16* k16  = (_Float16*)(ws + 135790592ull);     //  32 MB  f16 [32][1024][512]
    _Float16* vt16 = (_Float16*)(ws + 169345024ull);     //  32 MB  f16 [32][512][1024]
    float*    S    = (float*)(ws + 202899456ull);        // 128 MB  f32 [32][1024][1024]
    float*    o    = (float*)q16;                        //  64 MB  f32, reuses q+k region

    const long long LD = 1024ll * 512;   // tokens x dim per batch

    transpose_to_f16<<<dim3(1024, 3), 256, 0, stream>>>(Qw, Kw, Vw, Wt);
    conv1_relu<<<32768, 256, 0, stream>>>(x, W1, b1, h, hx);

    // q = hx . Q   (M=1024, N=512, K=512, per batch)
    gemm_wmma<_Float16><<<dim3(4, 8, 32), 256, 0, stream>>>(
        hx, Wt, q16, 512, 512, 512, 512, LD, 0, LD);
    // k = hx . K
    gemm_wmma<_Float16><<<dim3(4, 8, 32), 256, 0, stream>>>(
        hx, Wt + (1 << 18), k16, 512, 512, 512, 512, LD, 0, LD);
    // vT[n][l] = sum_d V[d][n] * hx[l][d]   (M=512, N=1024, K=512)
    gemm_wmma<_Float16><<<dim3(8, 4, 32), 256, 0, stream>>>(
        Wt + (2 << 18), hx, vt16, 512, 512, 512, 1024, 0, LD, LD);
    // S = q . k^T   (M=1024, N=1024, K=512), f32 out
    gemm_wmma<float><<<dim3(8, 8, 32), 256, 0, stream>>>(
        q16, k16, S, 512, 512, 512, 1024, LD, LD, 1024ll * 1024);
    // softmax rows; P (f16) written in place, row pitch 2048 halfs
    softmax_rows<<<32 * 1024, 256, 0, stream>>>(S);
    // o = P . v   (M=1024, N=512, K=1024), A row pitch 2048
    gemm_wmma<float><<<dim3(4, 8, 32), 256, 0, stream>>>(
        (const _Float16*)S, vt16, o, 1024, 2048, 1024, 512,
        2048ll * 1024, LD, LD);

    conv2_out<<<32768, 256, 0, stream>>>(h, o, W2, b2, outp);
}
